// SelectiveLinearAttention_39067022524923
// MI455X (gfx1250) — matrix operand order repaired and checked by reading, not compiled, not run on hardware
//
#include <hip/hip_runtime.h>

// ---------------- problem constants (from reference) ----------------
#define B_SZ     4
#define SEQ_L    4096
#define DHID     2048
#define NH       32
#define NSTATE   16
#define KCONV    4
#define DTRANK   128
#define D_INNER  512          // NH*NSTATE
#define P_DIM    1152         // DTRANK + 2*NH*NSTATE
#define M_TOK    (B_SZ*SEQ_L) // 16384
#define NCHUNK   32
#define LCHUNK   (SEQ_L/NCHUNK) // 128
#define NCHAIN   (B_SZ*D_INNER) // 2048
#define KC       32           // K-chunk per WMMA stage

typedef __attribute__((ext_vector_type(16))) __bf16 bf16x16;
typedef __attribute__((ext_vector_type(8)))  float  f32x8;

union Frag { bf16x16 v; uint4 q[2]; };

static __device__ __forceinline__ unsigned short f2bf(float f) {
    union { float f; unsigned u; } x; x.f = f;
    unsigned r = x.u + 0x7FFFu + ((x.u >> 16) & 1u);   // round-to-nearest-even
    return (unsigned short)(r >> 16);
}
static __device__ __forceinline__ float bf2f(unsigned short h) {
    union { unsigned u; float f; } x; x.u = ((unsigned)h) << 16;
    return x.f;
}
static __device__ __forceinline__ float silu(float v) {
    return v / (1.f + __expf(-v));
}

// ---- CDNA5 async global->LDS copy (ASYNCcnt path), with portable fallback ----
#if defined(__gfx1250__) && \
    __has_builtin(__builtin_amdgcn_global_load_async_to_lds_b128) && \
    __has_builtin(__builtin_amdgcn_s_wait_asynccnt)
#define HAVE_ASYNC_LDS 1
typedef int v4i __attribute__((vector_size(16)));
typedef __attribute__((address_space(1))) v4i gv4i;   // global int4
typedef __attribute__((address_space(3))) v4i lv4i;   // LDS int4
static __device__ __forceinline__ void async_cp16(const void* g, void* l) {
    __builtin_amdgcn_global_load_async_to_lds_b128((gv4i*)g, (lv4i*)l, 0, 0);
}
#else
#define HAVE_ASYNC_LDS 0
#endif

// ---------------- generic f32 -> bf16 conversion ----------------
__global__ void cvt_f32_bf16(const float* __restrict__ src,
                             unsigned short* __restrict__ dst, size_t n) {
    size_t t = (size_t)blockIdx.x * blockDim.x + threadIdx.x;
    if (t < n) dst[t] = f2bf(src[t]);
}

// ---------------- bf16 WMMA GEMM:  C(MxN) = A(MxK) * B(NxK)^T ----------------
// block = 8 waves; block tile 64(M) x 128(N); wave tile 32x32 (2x2 WMMA tiles)
// Double-buffered LDS staging with async global->LDS copies.
__global__ __launch_bounds__(256) void wmma_gemm_bf16(
        const unsigned short* __restrict__ A,   // M x K, row major, bf16 bits
        const unsigned short* __restrict__ Bw,  // N x K, row major, bf16 bits
        float* __restrict__ C,                  // M x N, row major, f32
        int M, int N, int K)
{
    __shared__ __align__(16) unsigned short lA[2][64 * KC];
    __shared__ __align__(16) unsigned short lB[2][128 * KC];

    const int m0 = blockIdx.y * 64;
    const int n0 = blockIdx.x * 128;
    const int t    = threadIdx.x;
    const int lane = t & 31;
    const int wave = t >> 5;
    const int wy = wave >> 2;      // 0..1  -> M
    const int wx = wave & 3;       // 0..3  -> N
    const int r  = lane & 15;
    const int kh = lane >> 4;      // lane group (K-half select)

    // staging slots for this thread (16B each)
    const int aRow = t >> 2,       aCol = (t & 3) << 3;          // A: 1x16B
    const int bRow0 = (t << 1) >> 2,       bCol0 = ((t << 1) & 3) << 3;  // B: 2x16B
    const int bRow1 = ((t << 1) + 1) >> 2, bCol1 = (((t << 1) + 1) & 3) << 3;

    f32x8 acc[2][2] = {};

#if HAVE_ASYNC_LDS
    // issue tile ki into buffer buf (3 async b128 copies per thread)
    #define ISSUE_TILE(k0, buf)                                                 \
        do {                                                                    \
            async_cp16(&A [(size_t)(m0 + aRow ) * K + (k0) + aCol ],            \
                       &lA[buf][aRow  * KC + aCol ]);                           \
            async_cp16(&Bw[(size_t)(n0 + bRow0) * K + (k0) + bCol0],            \
                       &lB[buf][bRow0 * KC + bCol0]);                           \
            async_cp16(&Bw[(size_t)(n0 + bRow1) * K + (k0) + bCol1],            \
                       &lB[buf][bRow1 * KC + bCol1]);                           \
        } while (0)
#else
    #define ISSUE_TILE(k0, buf)                                                 \
        do {                                                                    \
            uint4 ta = *(const uint4*)&A [(size_t)(m0 + aRow ) * K + (k0) + aCol ]; \
            uint4 tb = *(const uint4*)&Bw[(size_t)(n0 + bRow0) * K + (k0) + bCol0]; \
            uint4 tc = *(const uint4*)&Bw[(size_t)(n0 + bRow1) * K + (k0) + bCol1]; \
            *(uint4*)&lA[buf][aRow  * KC + aCol ] = ta;                         \
            *(uint4*)&lB[buf][bRow0 * KC + bCol0] = tb;                         \
            *(uint4*)&lB[buf][bRow1 * KC + bCol1] = tc;                         \
        } while (0)
#endif

    const int nk = K / KC;
    int buf = 0;
    ISSUE_TILE(0, 0);

    for (int ki = 0; ki < nk; ++ki) {
        // overlap: start next tile into the other buffer, then wait only for
        // the current tile's 3 async copies (in-order completion per wave).
        if (ki + 1 < nk) {
            ISSUE_TILE((ki + 1) * KC, buf ^ 1);
#if HAVE_ASYNC_LDS
            __builtin_amdgcn_s_wait_asynccnt(3);
#endif
        } else {
#if HAVE_ASYNC_LDS
            __builtin_amdgcn_s_wait_asynccnt(0);
#endif
        }
        __syncthreads();

        // ---- load fragments per CDNA5 16-bit WMMA VGPR layouts ----
        Frag fa[2], fb[2];
#pragma unroll
        for (int mt = 0; mt < 2; ++mt) {
            int row = wy * 32 + mt * 16 + r;
            // A 16x32: lanes<16 hold K{0..7,16..23}, lanes>=16 hold K{8..15,24..31}
            fa[mt].q[0] = *(const uint4*)&lA[buf][row * KC + kh * 8];
            fa[mt].q[1] = *(const uint4*)&lA[buf][row * KC + 16 + kh * 8];
        }
#pragma unroll
        for (int nt = 0; nt < 2; ++nt) {
            int col = wx * 32 + nt * 16 + r;
            // B 32x16: lanes<16 hold K{0..7,16..23}, lanes>=16 hold K{8..15,24..31}
            fb[nt].q[0] = *(const uint4*)&lB[buf][col * KC + kh * 8];
            fb[nt].q[1] = *(const uint4*)&lB[buf][col * KC + 16 + kh * 8];
        }
#pragma unroll
        for (int mt = 0; mt < 2; ++mt)
#pragma unroll
            for (int nt = 0; nt < 2; ++nt)
                acc[mt][nt] = __builtin_amdgcn_wmma_f32_16x16x32_bf16(
                    false, fa[mt].v, false, fb[nt].v,
                    (short)0, acc[mt][nt], false, false);

        __syncthreads();   // all waves done reading buf before it is refilled
        buf ^= 1;
    }
    #undef ISSUE_TILE

    // ---- epilogue: C/D 16x16 f32 layout: VGPR rr -> M=rr (+8 for lanes>=16), lane -> N
#pragma unroll
    for (int mt = 0; mt < 2; ++mt)
#pragma unroll
        for (int nt = 0; nt < 2; ++nt) {
            int gn = n0 + wx * 32 + nt * 16 + r;
#pragma unroll
            for (int rr = 0; rr < 8; ++rr) {
                int gm = m0 + wy * 32 + mt * 16 + rr + 8 * kh;
                C[(size_t)gm * N + gn] = acc[mt][nt][rr];
            }
        }
}

// ---------------- depthwise causal conv (K=4) + bias + SiLU -> bf16 ----------------
__global__ void conv_silu_bf16(const float* __restrict__ x,   // (B,L,D_INNER)
                               const float* __restrict__ cw,  // (D_INNER,1,K)
                               const float* __restrict__ cb,  // (D_INNER,)
                               unsigned short* __restrict__ xact)
{
    size_t t = (size_t)blockIdx.x * blockDim.x + threadIdx.x;
    size_t total = (size_t)B_SZ * SEQ_L * D_INNER;
    if (t >= total) return;
    int c = (int)(t % D_INNER);
    int l = (int)((t / D_INNER) % SEQ_L);
    int b = (int)(t / ((size_t)SEQ_L * D_INNER));
    float acc = cb[c];
#pragma unroll
    for (int k = 0; k < KCONV; ++k) {
        int ls = l - (KCONV - 1) + k;
        if (ls >= 0)
            acc += cw[c * KCONV + k] * x[((size_t)b * SEQ_L + ls) * D_INNER + c];
    }
    xact[t] = f2bf(silu(acc));
}

// ---------------- delta = softplus(dt_feats @ W_dt^T + b_dt) ----------------
__global__ void delta_kernel(const float* __restrict__ params, // (B*L, P_DIM)
                             const float* __restrict__ Wdt,    // (NH, DTRANK)
                             const float* __restrict__ bdt,    // (NH,)
                             float* __restrict__ delta)        // (B*L, NH)
{
    int t = blockIdx.x * blockDim.x + threadIdx.x;
    if (t >= M_TOK * NH) return;
    int h   = t & (NH - 1);
    int tok = t >> 5;
    const float* p = params + (size_t)tok * P_DIM;
    const float* w = Wdt + h * DTRANK;
    float s = bdt[h];
#pragma unroll 8
    for (int j = 0; j < DTRANK; ++j) s += p[j] * w[j];
    delta[t] = (s > 20.f) ? s : logf(1.f + __expf(s));
}

// ---------------- chunked linear-recurrence scan ----------------
// pass 1: per (chain, chunk) compute (prod a, local h); chain = b*512 + h*16 + n
__global__ void scan_pass1(const float* __restrict__ delta,
                           const float* __restrict__ params,
                           const unsigned short* __restrict__ xact,
                           const float* __restrict__ A_log,
                           float* __restrict__ carryA,
                           float* __restrict__ carryH)
{
    int t = blockIdx.x * blockDim.x + threadIdx.x;   // [0, NCHAIN*NCHUNK)
    int chain = t & (NCHAIN - 1);
    int chunk = t >> 11;
    int idx = chain & (D_INNER - 1);
    int b   = chain >> 9;
    int h   = idx >> 4;
    float Av = -__expf(A_log[idx]);
    float pa = 1.f, hl = 0.f;
    int l0 = chunk * LCHUNK;
    for (int s = 0; s < LCHUNK; ++s) {
        size_t tok = (size_t)b * SEQ_L + (l0 + s);
        float d  = delta[tok * NH + h];
        float a  = __expf(d * Av);
        float Br = params[tok * P_DIM + DTRANK + idx];
        float u  = bf2f(xact[tok * D_INNER + idx]);
        hl = a * hl + d * Br * u;
        pa *= a;
    }
    carryA[(size_t)chunk * NCHAIN + chain] = pa;
    carryH[(size_t)chunk * NCHAIN + chain] = hl;
}

// pass 2: sequential stitch over chunks (tiny: 2048 chains x 32 chunks)
__global__ void scan_pass2(const float* __restrict__ carryA,
                           const float* __restrict__ carryH,
                           float* __restrict__ hin)
{
    int chain = blockIdx.x * blockDim.x + threadIdx.x;
    if (chain >= NCHAIN) return;
    float hs = 0.f;
    for (int j = 0; j < NCHUNK; ++j) {
        hin[(size_t)j * NCHAIN + chain] = hs;
        hs = carryA[(size_t)j * NCHAIN + chain] * hs
           + carryH[(size_t)j * NCHAIN + chain];
    }
}

// pass 3: replay chunk from correct state; fuse y = C*h + D*x_act, gate silu(z), emit bf16
__global__ void scan_pass3(const float* __restrict__ delta,
                           const float* __restrict__ params,
                           const unsigned short* __restrict__ xact,
                           const float* __restrict__ zbuf,
                           const float* __restrict__ A_log,
                           const float* __restrict__ Dvec,
                           const float* __restrict__ hin,
                           unsigned short* __restrict__ ygate)
{
    int t = blockIdx.x * blockDim.x + threadIdx.x;
    int chain = t & (NCHAIN - 1);
    int chunk = t >> 11;
    int idx = chain & (D_INNER - 1);
    int b   = chain >> 9;
    int h   = idx >> 4;
    float Av = -__expf(A_log[idx]);
    float Dv = Dvec[idx];
    float hs = hin[(size_t)chunk * NCHAIN + chain];
    int l0 = chunk * LCHUNK;
    for (int s = 0; s < LCHUNK; ++s) {
        size_t tok = (size_t)b * SEQ_L + (l0 + s);
        float d  = delta[tok * NH + h];
        float a  = __expf(d * Av);
        float Br = params[tok * P_DIM + DTRANK + idx];
        float Cr = params[tok * P_DIM + DTRANK + D_INNER + idx];
        float u  = bf2f(xact[tok * D_INNER + idx]);
        hs = a * hs + d * Br * u;
        float y  = Cr * hs + Dv * u;
        float zv = zbuf[tok * D_INNER + idx];
        ygate[tok * D_INNER + idx] = f2bf(y * silu(zv));
    }
}

// ---------------- host side ----------------
extern "C" void kernel_launch(void* const* d_in, const int* in_sizes, int n_in,
                              void* d_out, int out_size, void* d_ws, size_t ws_size,
                              hipStream_t stream)
{
    (void)in_sizes; (void)n_in; (void)out_size; (void)ws_size;
    const float* hs    = (const float*)d_in[0];
    const float* Wx    = (const float*)d_in[1];
    const float* Wz    = (const float*)d_in[2];
    const float* convw = (const float*)d_in[3];
    const float* convb = (const float*)d_in[4];
    const float* Wp    = (const float*)d_in[5];
    const float* Wdt   = (const float*)d_in[6];
    const float* bdt   = (const float*)d_in[7];
    const float* Alog  = (const float*)d_in[8];
    const float* Dv    = (const float*)d_in[9];
    const float* Wo    = (const float*)d_in[10];
    float* out = (float*)d_out;

    char* ws = (char*)d_ws;
    size_t off = 0;
    auto carve = [&](size_t bytes) -> void* {
        void* p = ws + off;
        off += (bytes + 255) & ~(size_t)255;
        return p;
    };

    const size_t nHS   = (size_t)M_TOK * DHID;      // 33.6M
    const size_t nACT  = (size_t)M_TOK * D_INNER;   // 8.4M
    const size_t nPAR  = (size_t)M_TOK * P_DIM;     // 18.9M

    unsigned short* hsb  = (unsigned short*)carve(nHS * 2);
    unsigned short* wxb  = (unsigned short*)carve((size_t)D_INNER * DHID * 2);
    unsigned short* wzb  = (unsigned short*)carve((size_t)D_INNER * DHID * 2);
    unsigned short* wpb  = (unsigned short*)carve((size_t)P_DIM * D_INNER * 2);
    unsigned short* wob  = (unsigned short*)carve((size_t)DHID * D_INNER * 2);
    float*          xf   = (float*)carve(nACT * 4);
    float*          zf   = (float*)carve(nACT * 4);
    unsigned short* xact = (unsigned short*)carve(nACT * 2);
    float*          parf = (float*)carve(nPAR * 4);
    float*          dlt  = (float*)carve((size_t)M_TOK * NH * 4);
    float*          cA   = (float*)carve((size_t)NCHAIN * NCHUNK * 4);
    float*          cH   = (float*)carve((size_t)NCHAIN * NCHUNK * 4);
    float*          hin  = (float*)carve((size_t)NCHAIN * NCHUNK * 4);
    unsigned short* ygb  = (unsigned short*)carve(nACT * 2);

    const int T = 256;
    // 1) convert hidden_states + weights to bf16 (weights tiny; hs 67MB)
    cvt_f32_bf16<<<(unsigned)((nHS + T - 1) / T), T, 0, stream>>>(hs, hsb, nHS);
    cvt_f32_bf16<<<(unsigned)((D_INNER*(size_t)DHID + T - 1) / T), T, 0, stream>>>(Wx, wxb, (size_t)D_INNER * DHID);
    cvt_f32_bf16<<<(unsigned)((D_INNER*(size_t)DHID + T - 1) / T), T, 0, stream>>>(Wz, wzb, (size_t)D_INNER * DHID);
    cvt_f32_bf16<<<(unsigned)((P_DIM*(size_t)D_INNER + T - 1) / T), T, 0, stream>>>(Wp, wpb, (size_t)P_DIM * D_INNER);
    cvt_f32_bf16<<<(unsigned)((DHID*(size_t)D_INNER + T - 1) / T), T, 0, stream>>>(Wo, wob, (size_t)DHID * D_INNER);

    // 2) in-projections: x, z = hs @ W^T   (M=16384, N=512, K=2048)
    dim3 g1(D_INNER / 128, M_TOK / 64);
    wmma_gemm_bf16<<<g1, T, 0, stream>>>(hsb, wxb, xf, M_TOK, D_INNER, DHID);
    wmma_gemm_bf16<<<g1, T, 0, stream>>>(hsb, wzb, zf, M_TOK, D_INNER, DHID);

    // 3) depthwise causal conv + SiLU -> bf16 activations
    conv_silu_bf16<<<(unsigned)(nACT / T), T, 0, stream>>>(xf, convw, convb, xact);

    // 4) param projection: params = x_act @ W_param^T  (M=16384, N=1152, K=512)
    dim3 g2(P_DIM / 128, M_TOK / 64);
    wmma_gemm_bf16<<<g2, T, 0, stream>>>(xact, wpb, parf, M_TOK, P_DIM, D_INNER);

    // 5) delta
    delta_kernel<<<(M_TOK * NH) / T, T, 0, stream>>>(parf, Wdt, bdt, dlt);

    // 6) chunk-parallel scan (3 passes) with fused D-skip + silu(z) gate
    scan_pass1<<<(NCHAIN * NCHUNK) / T, T, 0, stream>>>(dlt, parf, xact, Alog, cA, cH);
    scan_pass2<<<NCHAIN / T, T, 0, stream>>>(cA, cH, hin);
    scan_pass3<<<(NCHAIN * NCHUNK) / T, T, 0, stream>>>(dlt, parf, xact, zf, Alog, Dv, hin, ygb);

    // 7) out-projection: out = y_gated @ W_out^T  (M=16384, N=2048, K=512)
    dim3 g3(DHID / 128, M_TOK / 64);
    wmma_gemm_bf16<<<g3, T, 0, stream>>>(ygb, wob, out, M_TOK, DHID, D_INNER);
}
